// Segment_Model_54992761258528
// MI455X (gfx1250) — matrix operand order repaired and checked by reading, not compiled, not run on hardware
//
#include <hip/hip_runtime.h>
#include <hip/hip_bf16.h>

// ---------------- model dimensions ----------------
#define TT   1024     // sequence length
#define HH   512      // hidden (decoder / concat)
#define H2   256      // per-direction hidden
#define G3   768      // 3*H2
#define D3   1536     // 3*HH
#define WE_  300
#define PE_  100
#define KPAD 512      // all GEMM K padded to 512
#define SMOO (-1e8f)

typedef __attribute__((ext_vector_type(16))) _Float16 v16h;
typedef __attribute__((ext_vector_type(8)))  float    v8f;

// ------------- pack fp32 weights -> fp16 B in WMMA *fragment order*, 64-col blocked --------
// Logical B is (K x N) (optionally src transposed), zero-padded. For 64-col block wb,
// k-tile kt (32 K) and sub-tile s (16 cols), lane L's 16 halves are contiguous, and the
// 4 sub-tiles of one (wb, kt) step form one contiguous 4KB chunk:
//   dst[(((wb*kTiles + kt)*4 + s)*32 + lane)*16 + j]
//     = B[kt*32 + (lane>>4)*16 + j][wb*64 + s*16 + (lane&15)]
__global__ void pack_b_kernel(const float* __restrict__ src, _Float16* __restrict__ dst,
                              int srcR, int srcC, int K, int N, int transpose) {
    int idx = blockIdx.x * blockDim.x + threadIdx.x;
    if (idx >= K * N) return;
    int kTiles = K >> 5;
    int j    = idx & 15;
    int lane = (idx >> 4) & 31;
    int frag = idx >> 9;
    int s    = frag & 3;
    int rem  = frag >> 2;
    int kt   = rem % kTiles;
    int wb   = rem / kTiles;
    int kk  = kt * 32 + (lane >> 4) * 16 + j;
    int col = wb * 64 + s * 16 + (lane & 15);
    float v = 0.f;
    if (transpose) {                       // B[k,n] = src[n,k]
        if (col < srcR && kk < srcC) v = src[(size_t)col * srcC + kk];
    } else {                               // B[k,n] = src[k,n]
        if (kk < srcR && col < srcC) v = src[(size_t)kk * srcC + col];
    }
    dst[idx] = (_Float16)v;
}

// ---------------- embedding gather -> f16 X (T x 512, zero padded past 400) ----------------
__global__ void embed_kernel(const int* __restrict__ word_ids, const int* __restrict__ pos_ids,
                             const int* __restrict__ masks,
                             const float* __restrict__ wemb, const float* __restrict__ pemb,
                             _Float16* __restrict__ xh) {
    int t = blockIdx.x;          // 0..T-1
    int c = threadIdx.x;         // 0..511
    float m = (float)masks[t];
    float v = 0.f;
    if (c < WE_)            v = wemb[(size_t)word_ids[t] * WE_ + c];
    else if (c < WE_ + PE_) v = pemb[(size_t)pos_ids[t] * PE_ + (c - WE_)];
    xh[(size_t)t * KPAD + c] = (_Float16)(v * m);
}

// ---------------- A-fragment load: 2x16B from one row (maps to 2x global_load_b128) --------
__device__ __forceinline__ v16h load_afrag(const _Float16* __restrict__ p) {
    v16h a;
    #pragma unroll
    for (int j = 0; j < 8; ++j) { a[j] = p[j]; a[8 + j] = p[8 + j]; }
    return a;
}

// ---------------- WMMA GEMM: C(MxN,f32) = A(MxK,f16) * Bpacked ----------------------------
// Each wave owns a 16(M) x 64(N) strip; double-buffered with unconditional prefetch and a
// peeled epilogue: next k-step's A + 4 B fragments (one clause, immediate offsets) stay in
// flight across the current step's 4 WMMAs.
__global__ void gemm_f16_wmma(const _Float16* __restrict__ A, const _Float16* __restrict__ Bp,
                              float* __restrict__ C, int M, int N, int K) {
    int wavesPerBlk = blockDim.x >> 5;
    int gw = blockIdx.x * wavesPerBlk + (threadIdx.x >> 5);
    int nBlk64 = N >> 6;
    int wm = gw / nBlk64;                 // 16-row tile index
    int wn = gw % nBlk64;                 // 64-col block index
    int tm = wm << 4;
    if (tm >= M) return;
    int lane = threadIdx.x & 31;
    int half = lane >> 4;
    int l16  = lane & 15;
    int kTiles = K >> 5;
    const _Float16* arow = A  + (size_t)(tm + l16) * K + half * 16;
    const _Float16* bp   = Bp + (size_t)wn * kTiles * 2048 + (size_t)lane * 16;
    v8f acc0 = {}, acc1 = {}, acc2 = {}, acc3 = {};
    // prologue: fragments for kt = 0
    v16h a  = load_afrag(arow);
    v16h b0 = *(const v16h*)(bp);
    v16h b1 = *(const v16h*)(bp + 512);
    v16h b2 = *(const v16h*)(bp + 1024);
    v16h b3 = *(const v16h*)(bp + 1536);
    for (int kt = 0; kt < kTiles - 1; ++kt) {
        // unconditional prefetch of next step (in flight across the 4 WMMAs below)
        v16h an = load_afrag(arow + 32);
        const _Float16* q = bp + 2048;
        v16h c0 = *(const v16h*)(q);
        v16h c1 = *(const v16h*)(q + 512);
        v16h c2 = *(const v16h*)(q + 1024);
        v16h c3 = *(const v16h*)(q + 1536);
        acc0 = __builtin_amdgcn_wmma_f32_16x16x32_f16(false, a, false, b0, (short)0, acc0, false, false);
        acc1 = __builtin_amdgcn_wmma_f32_16x16x32_f16(false, a, false, b1, (short)0, acc1, false, false);
        acc2 = __builtin_amdgcn_wmma_f32_16x16x32_f16(false, a, false, b2, (short)0, acc2, false, false);
        acc3 = __builtin_amdgcn_wmma_f32_16x16x32_f16(false, a, false, b3, (short)0, acc3, false, false);
        a = an; b0 = c0; b1 = c1; b2 = c2; b3 = c3;
        arow += 32;
        bp   += 2048;                     // 4 sub-tiles * 32 lanes * 16 halves
    }
    // epilogue: last k-step, no prefetch
    acc0 = __builtin_amdgcn_wmma_f32_16x16x32_f16(false, a, false, b0, (short)0, acc0, false, false);
    acc1 = __builtin_amdgcn_wmma_f32_16x16x32_f16(false, a, false, b1, (short)0, acc1, false, false);
    acc2 = __builtin_amdgcn_wmma_f32_16x16x32_f16(false, a, false, b2, (short)0, acc2, false, false);
    acc3 = __builtin_amdgcn_wmma_f32_16x16x32_f16(false, a, false, b3, (short)0, acc3, false, false);
    // C/D layout: VGPR r -> M = tm + half*8 + r, col = (wn*64) + s*16 + l16
    float* crow = C + (size_t)(tm + half * 8) * N + (wn << 6) + l16;
    #pragma unroll
    for (int r = 0; r < 8; ++r) {
        crow[0]  = acc0[r];
        crow[16] = acc1[r];
        crow[32] = acc2[r];
        crow[48] = acc3[r];
        crow += N;
    }
}

// ---------------- bi-GRU recurrence: block 0 = forward, block 1 = backward -----------------
__global__ void gru_scan_kernel(const float* __restrict__ gi_f, const float* __restrict__ gi_b,
                                const float* __restrict__ Whh_f, const float* __restrict__ Whh_b,
                                const float* __restrict__ bih_f, const float* __restrict__ bhh_f,
                                const float* __restrict__ bih_b, const float* __restrict__ bhh_b,
                                float* __restrict__ y, _Float16* __restrict__ yh,
                                float* __restrict__ hT) {
    __shared__ float h[H2];
    int dir = blockIdx.x;                 // 0 fwd, 1 bwd
    const float* gi = dir ? gi_b  : gi_f;
    const float* W  = dir ? Whh_b : Whh_f;
    const float* bi = dir ? bih_b : bih_f;
    const float* bh = dir ? bhh_b : bhh_f;
    int i = threadIdx.x;                  // 0..255
    h[i] = 0.f;
    __syncthreads();
    const float* w0 = W + (size_t)(i        ) * H2;
    const float* w1 = W + (size_t)(i + H2   ) * H2;
    const float* w2 = W + (size_t)(i + 2*H2 ) * H2;
    float bi0 = bi[i], bi1 = bi[i + H2], bi2 = bi[i + 2*H2];
    float bh0 = bh[i], bh1 = bh[i + H2], bh2 = bh[i + 2*H2];
    for (int s = 0; s < TT; ++s) {
        int t = dir ? (TT - 1 - s) : s;
        float hr = 0.f, hz = 0.f, hn = 0.f;
        for (int j = 0; j < H2; ++j) {
            float hj = h[j];
            hr += w0[j] * hj; hz += w1[j] * hj; hn += w2[j] * hj;
        }
        const float* g = gi + (size_t)t * G3;
        float r = 1.f / (1.f + expf(-((g[i]        + bi0) + (hr + bh0))));
        float z = 1.f / (1.f + expf(-((g[i + H2]   + bi1) + (hz + bh1))));
        float n = tanhf((g[i + 2*H2] + bi2) + r * (hn + bh2));
        float hnew = (1.f - z) * n + z * h[i];
        __syncthreads();
        h[i] = hnew;
        __syncthreads();
        int col = dir * H2 + i;
        y [(size_t)t * HH + col] = hnew;
        yh[(size_t)t * HH + col] = (_Float16)hnew;
        if (hT && s == TT - 1) hT[col] = hnew;
    }
}

// ---------------- he = elu(pre + be); write f32 + f16 copies -------------------------------
__global__ void elu_bias_kernel(const float* __restrict__ pre, const float* __restrict__ bias,
                                float* __restrict__ out, _Float16* __restrict__ outh, int total) {
    int idx = blockIdx.x * blockDim.x + threadIdx.x;
    if (idx >= total) return;
    int c = idx & (HH - 1);
    float v = pre[idx] + bias[c];
    v = v > 0.f ? v : (expf(v) - 1.f);
    out[idx]  = v;
    outh[idx] = (_Float16)v;
}

// ---------------- he_bias[t] = he[t,:] . we ------------------------------------------------
__global__ void rowdot_kernel(const float* __restrict__ he, const float* __restrict__ we,
                              float* __restrict__ out) {
    int t = blockIdx.x * blockDim.x + threadIdx.x;
    if (t >= TT) return;
    float s = 0.f;
    const float* row = he + (size_t)t * HH;
    for (int c = 0; c < HH; ++c) s += row[c] * we[c];
    out[t] = s;
}

// ---------------- sequential pointer decoder (single persistent workgroup) -----------------
__global__ void decoder_kernel(const float* __restrict__ gcn,  const float* __restrict__ heU,
                               const float* __restrict__ heb,  const float* __restrict__ henc,
                               const float* __restrict__ Wih,  const float* __restrict__ Whh,
                               const float* __restrict__ bih,  const float* __restrict__ bhh,
                               const float* __restrict__ Wd,   const float* __restrict__ bd,
                               const float* __restrict__ wd,   const float* __restrict__ bptr,
                               float* __restrict__ out) {
    __shared__ float hs[HH], dins[HH], hns[HH], gis[D3], ghs[D3], hds[HH], sc[TT], red[HH];
    __shared__ int   sstart, sdone, sbidx;
    int i = threadIdx.x;                  // 0..511
    hs[i] = henc[i];
    if (i == 0) { sstart = 0; sdone = 0; }
    float bconst = *bptr;
    __syncthreads();
    for (int step = 0; step < TT; ++step) {
        if (sdone) {                      // uniform branch: carry frozen, outputs -1 / zeros
            if (i == 0) out[step] = -1.0f;
            out[TT + (size_t)step * TT + i]       = 0.f;
            out[TT + (size_t)step * TT + i + HH]  = 0.f;
            __syncthreads();
            continue;
        }
        int start = sstart;
        dins[i] = gcn[(size_t)start * HH + i];
        __syncthreads();
        // gi = Wih@din + bih ; gh = Whh@h + bhh  (each thread: rows i, i+512, i+1024)
        #pragma unroll
        for (int r = 0; r < 3; ++r) {
            int row = i + r * HH;
            const float* wi = Wih + (size_t)row * HH;
            const float* wh = Whh + (size_t)row * HH;
            float si = 0.f, sh = 0.f;
            for (int j = 0; j < HH; ++j) { si += wi[j] * dins[j]; sh += wh[j] * hs[j]; }
            gis[row] = si + bih[row];
            ghs[row] = sh + bhh[row];
        }
        __syncthreads();
        {   // GRU cell update
            float r = 1.f / (1.f + expf(-(gis[i] + ghs[i])));
            float z = 1.f / (1.f + expf(-(gis[HH + i] + ghs[HH + i])));
            float n = tanhf(gis[2 * HH + i] + r * ghs[2 * HH + i]);
            hns[i] = (1.f - z) * n + z * hs[i];
        }
        __syncthreads();
        {   // hd = elu(hnew @ Wd + bd)   (column dot, coalesced over threads)
            float s = 0.f;
            for (int j = 0; j < HH; ++j) s += hns[j] * Wd[(size_t)j * HH + i];
            s += bd[i];
            hds[i] = s > 0.f ? s : (expf(s) - 1.f);
        }
        __syncthreads();
        // wd_dot = hd . wd  (+ scalar bias)
        red[i] = hds[i] * wd[i];
        __syncthreads();
        for (int off = 256; off > 0; off >>= 1) {
            if (i < off) red[i] += red[i + off];
            __syncthreads();
        }
        float wdd = red[0] + bconst;
        // masked biaffine scores: each thread owns rows t = i, i+512
        #pragma unroll
        for (int r = 0; r < 2; ++r) {
            int t = i + r * HH;
            const float* hu = heU + (size_t)t * HH;
            float s = 0.f;
            for (int j = 0; j < HH; ++j) s += hu[j] * hds[j];
            s += heb[t] + wdd;
            sc[t] = (t >= start) ? s : SMOO;
        }
        __syncthreads();
        // block max
        red[i] = fmaxf(sc[i], sc[i + HH]);
        __syncthreads();
        for (int off = 256; off > 0; off >>= 1) {
            if (i < off) red[i] = fmaxf(red[i], red[i + off]);
            __syncthreads();
        }
        float mx = red[0];
        if (i == 0) sbidx = TT;
        __syncthreads();
        // first-index argmax (jnp.argmax semantics)
        if (sc[i]      == mx) atomicMin(&sbidx, i);
        if (sc[i + HH] == mx) atomicMin(&sbidx, i + HH);
        // log-sum-exp
        red[i] = expf(sc[i] - mx) + expf(sc[i + HH] - mx);
        __syncthreads();
        for (int off = 256; off > 0; off >>= 1) {
            if (i < off) red[i] += red[i + off];
            __syncthreads();
        }
        float lse = mx + logf(red[0]);
        int bidx = sbidx;
        // outputs
        if (i == 0) out[step] = (float)bidx;
        out[TT + (size_t)step * TT + i]      = sc[i]      - lse;
        out[TT + (size_t)step * TT + i + HH] = sc[i + HH] - lse;
        __syncthreads();
        // carry update
        hs[i] = hns[i];
        if (i == 0) {
            int ns = bidx + 1;
            sstart = ns;
            if (ns >= TT) sdone = 1;
        }
        __syncthreads();
    }
}

// =================================== host orchestration ===================================
extern "C" void kernel_launch(void* const* d_in, const int* in_sizes, int n_in,
                              void* d_out, int out_size, void* d_ws, size_t ws_size,
                              hipStream_t stream) {
    (void)in_sizes; (void)n_in; (void)out_size; (void)ws_size;
    // ---- inputs (setup_inputs insertion order, params flattened recursively) ----
    const int*   word_ids = (const int*)d_in[0];
    const int*   pos_ids  = (const int*)d_in[1];
    const int*   masks    = (const int*)d_in[2];
    const float* wemb     = (const float*)d_in[3];
    const float* pemb     = (const float*)d_in[4];
    // GRU param blocks: {Wih, Whh, bih, bhh}; dirs: s0.f(5) s0.b(9) s1.f(13) s1.b(17) enc.f(21) enc.b(25)
    const int gruBase[6] = {5, 9, 13, 17, 21, 25};
    // dec at 29..32 ; ptr: We 33, be 34, Wd 35, bd 36, U 37, we 38, wd 39, b 40
    const float* decWih = (const float*)d_in[29];
    const float* decWhh = (const float*)d_in[30];
    const float* decbih = (const float*)d_in[31];
    const float* decbhh = (const float*)d_in[32];
    const float* ptrWe  = (const float*)d_in[33];
    const float* ptrbe  = (const float*)d_in[34];
    const float* ptrWd  = (const float*)d_in[35];
    const float* ptrbd  = (const float*)d_in[36];
    const float* ptrU   = (const float*)d_in[37];
    const float* ptrwe  = (const float*)d_in[38];
    const float* ptrwd  = (const float*)d_in[39];
    const float* ptrb   = (const float*)d_in[40];

    // ---- workspace bump allocator ----
    size_t off = 0;
    auto alloc = [&](size_t bytes) -> void* {
        off = (off + 255) & ~(size_t)255;
        void* p = (char*)d_ws + off;
        off += bytes;
        return p;
    };
    _Float16* xhA   = (_Float16*)alloc((size_t)TT * KPAD * 2);
    _Float16* xhB   = (_Float16*)alloc((size_t)TT * KPAD * 2);
    float*    ybuf  = (float*)   alloc((size_t)TT * HH * 4);    // ends as gcn
    float*    ebuf  = (float*)   alloc((size_t)TT * HH * 4);    // enc f32 output
    float*    gif   = (float*)   alloc((size_t)TT * G3 * 4);    // reused as he_pre
    float*    gib   = (float*)   alloc((size_t)TT * G3 * 4);    // reused as heU
    _Float16* BihP[6];
    for (int d = 0; d < 6; ++d) BihP[d] = (_Float16*)alloc((size_t)KPAD * G3 * 2);
    _Float16* WeB   = (_Float16*)alloc((size_t)KPAD * HH * 2);
    _Float16* UtB   = (_Float16*)alloc((size_t)KPAD * HH * 2);
    float*    he    = (float*)   alloc((size_t)TT * HH * 4);
    _Float16* heH   = (_Float16*)alloc((size_t)TT * HH * 2);
    float*    hebias= (float*)   alloc((size_t)TT * 4);
    float*    henc  = (float*)   alloc((size_t)HH * 4);

    // ---- pack all GEMM B operands to f16 fragment order ----
    for (int d = 0; d < 6; ++d) {
        int din = (d < 2) ? (WE_ + PE_) : HH;               // layer0: 400, else 512
        const float* Wih = (const float*)d_in[gruBase[d]];
        int total = KPAD * G3;
        pack_b_kernel<<<(total + 255) / 256, 256, 0, stream>>>(Wih, BihP[d], G3, din, KPAD, G3, 1);
    }
    pack_b_kernel<<<(KPAD * HH + 255) / 256, 256, 0, stream>>>(ptrWe, WeB, HH, HH, KPAD, HH, 0);
    pack_b_kernel<<<(KPAD * HH + 255) / 256, 256, 0, stream>>>(ptrU,  UtB, HH, HH, KPAD, HH, 1);

    // ---- embeddings ----
    embed_kernel<<<TT, KPAD, 0, stream>>>(word_ids, pos_ids, masks, wemb, pemb, xhA);

    // ---- 3 bi-GRU layers (sent0, sent1, enc) ----
    _Float16* xcur = xhA;
    for (int li = 0; li < 3; ++li) {
        int fB = gruBase[2 * li], bB = gruBase[2 * li + 1];
        const float* Whh_f = (const float*)d_in[fB + 1];
        const float* bih_f = (const float*)d_in[fB + 2];
        const float* bhh_f = (const float*)d_in[fB + 3];
        const float* Whh_b = (const float*)d_in[bB + 1];
        const float* bih_b = (const float*)d_in[bB + 2];
        const float* bhh_b = (const float*)d_in[bB + 3];
        int waves = (TT / 16) * (G3 / 64);                   // 64 * 12 = 768 waves
        gemm_f16_wmma<<<waves / 4, 128, 0, stream>>>(xcur, BihP[2 * li],     gif, TT, G3, KPAD);
        gemm_f16_wmma<<<waves / 4, 128, 0, stream>>>(xcur, BihP[2 * li + 1], gib, TT, G3, KPAD);
        float*    yout = (li == 2) ? ebuf : ybuf;            // ybuf ends holding gcn (sent1)
        _Float16* yh   = (xcur == xhA) ? xhB : xhA;
        float*    hT   = (li == 2) ? henc : (float*)nullptr;
        gru_scan_kernel<<<2, H2, 0, stream>>>(gif, gib, Whh_f, Whh_b,
                                              bih_f, bhh_f, bih_b, bhh_b, yout, yh, hT);
        xcur = yh;                                           // f16 output feeds next layer
    }

    // ---- pointer precompute: he, he_bias, heU ----
    float* hepre = gif;                                      // reuse (free after scans)
    float* heU   = gib;                                      // reuse
    int wavesH = (TT / 16) * (HH / 64);                      // 64 * 8 = 512 waves
    gemm_f16_wmma<<<wavesH / 4, 128, 0, stream>>>(xcur, WeB, hepre, TT, HH, KPAD);
    elu_bias_kernel<<<(TT * HH + 255) / 256, 256, 0, stream>>>(hepre, ptrbe, he, heH, TT * HH);
    rowdot_kernel<<<(TT + 255) / 256, 256, 0, stream>>>(he, ptrwe, hebias);
    gemm_f16_wmma<<<wavesH / 4, 128, 0, stream>>>(heH, UtB, heU, TT, HH, KPAD);

    // ---- sequential pointer decoder ----
    decoder_kernel<<<1, HH, 0, stream>>>(ybuf, heU, hebias, henc,
                                         decWih, decWhh, decbih, decbhh,
                                         ptrWd, ptrbd, ptrwd, ptrb, (float*)d_out);
}
